// CausalSelfAttention_28269474742974
// MI455X (gfx1250) — matrix operand order repaired and checked by reading, not compiled, not run on hardware
//
#include <hip/hip_runtime.h>
#include <hip/hip_bf16.h>
#include <math.h>

// ---------------------------------------------------------------------------
// CDNA5 (gfx1250) causal self-attention block: RMSNorm -> QKV GEMM -> RoPE ->
// flash causal attention (TDM-staged K/V tiles) -> projection.
// All matmuls via v_wmma_f32_16x16x32_f16; K/V chunks staged to LDS with the
// Tensor Data Mover (tensor_load_to_lds + s_wait_tensorcnt) when available.
// ---------------------------------------------------------------------------

typedef __attribute__((ext_vector_type(16))) _Float16 v16h;
typedef __attribute__((ext_vector_type(8)))  _Float16 v8h;
typedef __attribute__((ext_vector_type(8)))  float    v8f;
typedef __attribute__((ext_vector_type(4)))  unsigned int u32x4;
typedef __attribute__((ext_vector_type(8)))  int       i32x8;
typedef __attribute__((ext_vector_type(4)))  int       i32x4;

#define D_MODEL   1024
#define QKV_DIM   3072
#define NUM_HEADS 16
#define HEAD_DIM  64
#define SEQ       2048
#define BATCH     2
#define M_TOTAL   (BATCH * SEQ)   // 4096

#if defined(__gfx1250__) && __has_builtin(__builtin_amdgcn_tensor_load_to_lds) && \
    __has_builtin(__builtin_amdgcn_s_wait_tensorcnt)
#define HAS_TDM 1
#else
#define HAS_TDM 0
#endif

// ---------------------------------------------------------------------------
// WMMA helpers (wave32; A and B use contiguous k = 16*(lane/16) + i)
// ---------------------------------------------------------------------------

__device__ __forceinline__ v8f wmma_f16(v16h a, v16h b, v8f c) {
  // D = A(16x32 f16) * B(32x16 f16) + C(16x16 f32)
  return __builtin_amdgcn_wmma_f32_16x16x32_f16(
      /*neg_a=*/false, a, /*neg_b=*/false, b,
      /*c_mod=*/(short)0, c, /*reuse_a=*/false, /*reuse_b=*/false);
}

// A fragment: 16x32 f16, row-major source with leading dim `ld` (elements).
__device__ __forceinline__ v16h ldA(const _Float16* tileBase, int ld) {
  const int lane = threadIdx.x & 31;
  const int row  = lane & 15;
  const int kb   = (lane >> 4) * 16;
  const _Float16* p = tileBase + (size_t)row * ld + kb;
  v8h lo = *(const v8h*)(p);        // K = kb + 0..7
  v8h hi = *(const v8h*)(p + 8);    // K = kb + 8..15
  v16h a;
#pragma unroll
  for (int i = 0; i < 8; ++i) { a[i] = lo[i]; a[8 + i] = hi[i]; }
  return a;
}

// B fragment: 32x16 f16 where B[k][n] = rowMajorSrc[n][k] (i.e. B = srcT).
__device__ __forceinline__ v16h ldBT(const _Float16* tileBase, int ld) {
  const int lane = threadIdx.x & 31;
  const int n    = lane & 15;
  const int kb   = (lane >> 4) * 16;
  const _Float16* p = tileBase + (size_t)n * ld + kb;
  return *(const v16h*)p;
}

// B fragment: 32x16 f16 where B[k][n] = colBase[k*ld + n] (row-major K x N).
__device__ __forceinline__ v16h ldB_rows(const _Float16* colBase, int ld) {
  const int lane = threadIdx.x & 31;
  const int n    = lane & 15;
  const int kb   = (lane >> 4) * 16;
  v16h b;
#pragma unroll
  for (int i = 0; i < 16; ++i) b[i] = colBase[(size_t)(kb + i) * ld + n];
  return b;
}

__device__ __forceinline__ float rowmax16(float v) {
#pragma unroll
  for (int m = 8; m >= 1; m >>= 1) v = fmaxf(v, __shfl_xor(v, m, 16));
  return v;
}
__device__ __forceinline__ float rowsum16(float v) {
#pragma unroll
  for (int m = 8; m >= 1; m >>= 1) v += __shfl_xor(v, m, 16);
  return v;
}

// ---------------------------------------------------------------------------
// TDM: stage a rows x cols f16 tile (row stride srcStride elements) into LDS.
// Builds a 2-D Tensor DMA Descriptor per CDNA5 ISA 8.3/8.4:
//   group0: count=1 | lds_addr | global_addr[56:0] | type=2
//   group1: data_size=2B | tensor_dim0=cols | tensor_dim1=rows
//           tile_dim0=cols | tile_dim1=rows | tensor_dim0_stride=srcStride
// Fallback (toolchains without the builtin): cooperative coalesced staging.
// ---------------------------------------------------------------------------
__device__ __forceinline__ void stage_tile_to_lds(const _Float16* gsrc,
                                                  _Float16* ldsDst,
                                                  int rows, int cols,
                                                  int srcStride) {
#if HAS_TDM
  const unsigned long long ga = (unsigned long long)(uintptr_t)gsrc;
  const unsigned ldsOff = (unsigned)(uintptr_t)ldsDst;
  u32x4 g0;
  g0[0] = 1u;                                   // count=1 (valid user D#)
  g0[1] = ldsOff;                               // lds_addr (bytes)
  g0[2] = (unsigned)(ga & 0xffffffffu);         // global_addr[31:0]
  g0[3] = (unsigned)((ga >> 32) & 0x01ffffffu)  // global_addr[56:32]
        | (2u << 30);                           // type = 2 ("image")
  i32x8 g1;
  g1[0] = (1 << 16);                            // data_size=1 (2B); no mcast
  g1[1] = (cols & 0xffff) << 16;                // tensor_dim0[15:0]
  g1[2] = ((unsigned)cols >> 16) | ((rows & 0xffff) << 16); // dim0 hi | dim1 lo
  g1[3] = ((unsigned)rows >> 16) | ((cols & 0xffff) << 16); // dim1 hi | tile_dim0
  g1[4] = (rows & 0xffff);                      // tile_dim1 | tile_dim2=0
  g1[5] = srcStride;                            // tensor_dim0_stride[31:0]
  g1[6] = 0;                                    // stride hi | dim1_stride lo
  g1[7] = 0;
  const i32x4 z4 = {0, 0, 0, 0};
#if defined(__clang_major__) && (__clang_major__ >= 23)
  const i32x8 z8 = {0, 0, 0, 0, 0, 0, 0, 0};
  __builtin_amdgcn_tensor_load_to_lds(g0, g1, z4, z4, z8, 0);
#else
  __builtin_amdgcn_tensor_load_to_lds(g0, g1, z4, z4, 0);
#endif
#else
  // cooperative coalesced staging (one wave): 4B per lane per iteration
  const int lane = threadIdx.x & 31;
  unsigned* d = (unsigned*)ldsDst;
  const int total2 = (rows * cols) >> 1;        // # of 4-byte words
  const int cols2 = cols >> 1;
  for (int i = lane; i < total2; i += 32) {
    const int r = i / cols2;
    const int cp = i - r * cols2;
    d[i] = *(const unsigned*)(gsrc + (size_t)r * srcStride + cp * 2);
  }
#endif
}

__device__ __forceinline__ void stage_wait() {
#if HAS_TDM
  __builtin_amdgcn_s_wait_tensorcnt(0);
#endif
  __syncthreads();
}

// ---------------------------------------------------------------------------
// Kernel 0: f32 -> f16 convert (weights)
// ---------------------------------------------------------------------------
__global__ void cvt_f32_f16_kernel(const float* __restrict__ in,
                                   _Float16* __restrict__ out, int n) {
  int i = blockIdx.x * blockDim.x + threadIdx.x;
  if (i < n) out[i] = (_Float16)in[i];
}

// ---------------------------------------------------------------------------
// Kernel 1: RMSNorm row -> f16
// ---------------------------------------------------------------------------
__global__ void rmsnorm_kernel(const float* __restrict__ x,
                               const float* __restrict__ w,
                               _Float16* __restrict__ xn) {
  const int row = blockIdx.x;
  const float* xr = x + (size_t)row * D_MODEL;
  float ss = 0.f;
#pragma unroll
  for (int i = 0; i < 4; ++i) {
    float v = xr[threadIdx.x + 256 * i];
    ss += v * v;
  }
  __shared__ float red[256];
  red[threadIdx.x] = ss;
  __syncthreads();
#pragma unroll
  for (int s = 128; s > 0; s >>= 1) {
    if (threadIdx.x < s) red[threadIdx.x] += red[threadIdx.x + s];
    __syncthreads();
  }
  const float inv = rsqrtf(red[0] * (1.0f / D_MODEL) + 1e-6f);
  _Float16* o = xn + (size_t)row * D_MODEL;
#pragma unroll
  for (int i = 0; i < 4; ++i) {
    int idx = threadIdx.x + 256 * i;
    o[idx] = (_Float16)(xr[idx] * inv * w[idx]);
  }
}

// ---------------------------------------------------------------------------
// Kernel 2/5: GEMM  C[M,N] = A[M,K] * W[N,K]^T + bias   (one 16x16 tile/wave)
// ---------------------------------------------------------------------------
__global__ void gemm_wmma_kernel(const _Float16* __restrict__ A,
                                 const _Float16* __restrict__ W,
                                 const float* __restrict__ bias,
                                 _Float16* __restrict__ outH,
                                 float* __restrict__ outF,
                                 int M, int N, int K) {
  const int wavesPerBlock = blockDim.x >> 5;
  const int tile = blockIdx.x * wavesPerBlock + (threadIdx.x >> 5);
  const int tilesN = N >> 4;
  const int tm = tile / tilesN;
  const int tn = tile - tm * tilesN;
  if (tm * 16 >= M) return;  // wave-uniform
  const _Float16* Ab = A + (size_t)tm * 16 * K;
  const _Float16* Wb = W + (size_t)tn * 16 * K;
  v8f c = {0.f, 0.f, 0.f, 0.f, 0.f, 0.f, 0.f, 0.f};
  for (int k0 = 0; k0 < K; k0 += 32) {
    __builtin_prefetch(Ab + k0 + 512, 0, 1);  // global_prefetch_b8
    v16h a = ldA(Ab + k0, K);
    v16h b = ldBT(Wb + k0, K);
    c = wmma_f16(a, b, c);
  }
  const int lane = threadIdx.x & 31;
  const int n = lane & 15;
  const int g = lane >> 4;
  const int col = tn * 16 + n;
  const float bi = bias[col];
#pragma unroll
  for (int r = 0; r < 8; ++r) {
    const size_t row = (size_t)tm * 16 + r + 8 * g;
    const float v = c[r] + bi;
    if (outF) outF[row * N + col] = v;
    else      outH[row * N + col] = (_Float16)v;
  }
}

// ---------------------------------------------------------------------------
// Kernel 3: RoPE in-place on q and k halves of qkv (f16), head_dim = 64.
// ---------------------------------------------------------------------------
__global__ void rope_kernel(_Float16* __restrict__ qkv) {
  const int i  = blockIdx.x * blockDim.x + threadIdx.x;  // < 2^21
  const int dd = i & 31;
  const int hh = (i >> 5) & (NUM_HEADS - 1);
  const int ss = (i >> 9) & (SEQ - 1);
  const int bb = i >> 20;
  if (bb >= BATCH) return;
  // inv_freq = base^(-dd/32) ; ln(10000) = 9.210340371976184
  const float ang = (float)ss * __expf(-((float)dd * (1.0f / 32.0f)) * 9.210340371976184f);
  const float cs = __cosf(ang), sn = __sinf(ang);
  _Float16* q = qkv + ((size_t)(bb * SEQ + ss)) * QKV_DIM + hh * HEAD_DIM;
  _Float16* k = q + D_MODEL;
  const float q1 = (float)q[dd], q2 = (float)q[dd + 32];
  q[dd]      = (_Float16)(q1 * cs - q2 * sn);
  q[dd + 32] = (_Float16)(q2 * cs + q1 * sn);
  const float k1 = (float)k[dd], k2 = (float)k[dd + 32];
  k[dd]      = (_Float16)(k1 * cs - k2 * sn);
  k[dd + 32] = (_Float16)(k2 * cs + k1 * sn);
}

// ---------------------------------------------------------------------------
// Kernel 4: flash causal attention, one wave per (b, h, 16-query tile).
// K/V 32x64 chunks staged to LDS via TDM; S and O matmuls on WMMA.
// ---------------------------------------------------------------------------
__global__ void attn_kernel(const _Float16* __restrict__ qkv,
                            _Float16* __restrict__ o) {
  const int bid = blockIdx.x;
  const int qt = bid & 127;          // query tile (of 16 rows)
  const int h  = (bid >> 7) & (NUM_HEADS - 1);
  const int bb = bid >> 11;
  const int lane = threadIdx.x & 31;
  const int n = lane & 15;
  const int g = lane >> 4;

  __shared__ __align__(32)  _Float16 pbuf[16 * 32];   // P tile (C->A relayout)
  __shared__ __align__(128) _Float16 kbuf[32 * 64];   // K chunk, row-major
  __shared__ __align__(128) _Float16 vbuf[32 * 64];   // V chunk, row-major

  const _Float16* base = qkv + (size_t)bb * SEQ * QKV_DIM;
  const _Float16* Qp = base + (size_t)h * HEAD_DIM;
  const _Float16* Kp = base + D_MODEL + (size_t)h * HEAD_DIM;
  const _Float16* Vp = base + 2 * D_MODEL + (size_t)h * HEAD_DIM;

  // Q tile, head_dim split into two K=32 chunks (held in registers)
  const v16h qf0 = ldA(Qp + (size_t)(qt * 16) * QKV_DIM + 0,  QKV_DIM);
  const v16h qf1 = ldA(Qp + (size_t)(qt * 16) * QKV_DIM + 32, QKV_DIM);

  v8f acc0 = {0.f,0.f,0.f,0.f,0.f,0.f,0.f,0.f};
  v8f acc1 = acc0, acc2 = acc0, acc3 = acc0;
  float mrow[8], lrow[8];
#pragma unroll
  for (int r = 0; r < 8; ++r) { mrow[r] = -1e30f; lrow[r] = 0.f; }

  const int kEnd = qt * 16 + 16;
  const float scale = 0.125f;  // 1/sqrt(64)

  for (int k0 = 0; k0 < kEnd; k0 += 32) {
    // ---- TDM: stage K and V 32x64 chunks into LDS ------------------------
    stage_tile_to_lds(Kp + (size_t)k0 * QKV_DIM, kbuf, 32, 64, QKV_DIM);
    stage_tile_to_lds(Vp + (size_t)k0 * QKV_DIM, vbuf, 32, 64, QKV_DIM);
    stage_wait();  // s_wait_tensorcnt 0 (+ barrier)

    // ---- S = scale * Q K^T for key columns [k0, k0+32) -------------------
    v8f s0 = {0.f,0.f,0.f,0.f,0.f,0.f,0.f,0.f};
    v8f s1 = s0;
    s0 = wmma_f16(qf0, ldBT(kbuf + 0,           64), s0);
    s0 = wmma_f16(qf1, ldBT(kbuf + 32,          64), s0);
    s1 = wmma_f16(qf0, ldBT(kbuf + 16 * 64,      64), s1);
    s1 = wmma_f16(qf1, ldBT(kbuf + 16 * 64 + 32, 64), s1);

    // ---- causal mask + online softmax -----------------------------------
    float corr[8];
#pragma unroll
    for (int r = 0; r < 8; ++r) {
      const int grow = qt * 16 + r + 8 * g;
      float v0 = s0[r] * scale;
      float v1 = s1[r] * scale;
      if (k0 + n > grow)      v0 = -1e30f;
      if (k0 + 16 + n > grow) v1 = -1e30f;
      float rm = rowmax16(fmaxf(v0, v1));
      const float nm = fmaxf(mrow[r], rm);
      const float p0 = __expf(v0 - nm);
      const float p1 = __expf(v1 - nm);
      const float rs = rowsum16(p0 + p1);
      const float cc = __expf(mrow[r] - nm);
      lrow[r] = lrow[r] * cc + rs;
      mrow[r] = nm;
      corr[r] = cc;
      pbuf[(r + 8 * g) * 32 + n]      = (_Float16)p0;
      pbuf[(r + 8 * g) * 32 + 16 + n] = (_Float16)p1;
    }
#pragma unroll
    for (int r = 0; r < 8; ++r) {
      acc0[r] *= corr[r]; acc1[r] *= corr[r];
      acc2[r] *= corr[r]; acc3[r] *= corr[r];
    }
    __syncthreads();
    const v16h pa = ldA(pbuf, 32);   // P tile 16x32 as A fragment

    // ---- O += P * V chunk (from LDS) -------------------------------------
    acc0 = wmma_f16(pa, ldB_rows(vbuf + 0,  64), acc0);
    acc1 = wmma_f16(pa, ldB_rows(vbuf + 16, 64), acc1);
    acc2 = wmma_f16(pa, ldB_rows(vbuf + 32, 64), acc2);
    acc3 = wmma_f16(pa, ldB_rows(vbuf + 48, 64), acc3);
    __syncthreads();  // WAR: protect kbuf/vbuf/pbuf before next chunk's DMA
  }

  // ---- epilogue: normalize and store [b, s, h*64 + n] as f16 -------------
#pragma unroll
  for (int r = 0; r < 8; ++r) {
    const float inv = 1.0f / lrow[r];
    const int row = qt * 16 + r + 8 * g;
    _Float16* op = o + ((size_t)(bb * SEQ + row)) * D_MODEL + h * HEAD_DIM;
    op[n]      = (_Float16)(acc0[r] * inv);
    op[16 + n] = (_Float16)(acc1[r] * inv);
    op[32 + n] = (_Float16)(acc2[r] * inv);
    op[48 + n] = (_Float16)(acc3[r] * inv);
  }
}

// ---------------------------------------------------------------------------
// Launch
// ---------------------------------------------------------------------------
extern "C" void kernel_launch(void* const* d_in, const int* in_sizes, int n_in,
                              void* d_out, int out_size, void* d_ws, size_t ws_size,
                              hipStream_t stream) {
  (void)in_sizes; (void)n_in; (void)out_size; (void)ws_size;
  const float* x      = (const float*)d_in[0];
  const float* norm_w = (const float*)d_in[1];
  const float* qkv_w  = (const float*)d_in[2];
  const float* qkv_b  = (const float*)d_in[3];
  const float* proj_w = (const float*)d_in[4];
  const float* proj_b = (const float*)d_in[5];
  float* out = (float*)d_out;

  char* ws = (char*)d_ws;
  _Float16* xn    = (_Float16*)(ws);                        //  8 MB [4096,1024]
  _Float16* qkv   = (_Float16*)(ws + ((size_t)8  << 20));   // 24 MB [4096,3072]
  _Float16* wqkv  = (_Float16*)(ws + ((size_t)32 << 20));   //  6 MB [3072,1024]
  _Float16* wproj = (_Float16*)(ws + ((size_t)38 << 20));   //  2 MB [1024,1024]
  _Float16* oh    = (_Float16*)(ws + ((size_t)40 << 20));   //  8 MB [4096,1024]

  // 0) weight conversion to f16
  cvt_f32_f16_kernel<<<(QKV_DIM * D_MODEL) / 256, 256, 0, stream>>>(qkv_w, wqkv, QKV_DIM * D_MODEL);
  cvt_f32_f16_kernel<<<(D_MODEL * D_MODEL) / 256, 256, 0, stream>>>(proj_w, wproj, D_MODEL * D_MODEL);
  // 1) RMSNorm
  rmsnorm_kernel<<<M_TOTAL, 256, 0, stream>>>(x, norm_w, xn);
  // 2) fused QKV GEMM: [4096,3072] = xn * qkv_w^T + b
  gemm_wmma_kernel<<<6144, 256, 0, stream>>>(xn, wqkv, qkv_b, qkv, nullptr,
                                             M_TOTAL, QKV_DIM, D_MODEL);
  // 3) RoPE on q,k
  rope_kernel<<<8192, 256, 0, stream>>>(qkv);
  // 4) flash causal attention: one wave per (b,h,qtile)
  attn_kernel<<<BATCH * NUM_HEADS * (SEQ / 16), 32, 0, stream>>>(qkv, oh);
  // 5) projection: out[4096,1024] f32 = oh * proj_w^T + b
  gemm_wmma_kernel<<<2048, 256, 0, stream>>>(oh, wproj, proj_b, nullptr, out,
                                             M_TOTAL, D_MODEL, D_MODEL);
}
